// SparseSpikeFullAttention_18038862643389
// MI455X (gfx1250) — compile-verified, matrix operand order checked by reading, not run on hardware
//
#include <hip/hip_runtime.h>
#include <hip/hip_bf16.h>
#include <math.h>

typedef __attribute__((ext_vector_type(16))) _Float16 v16h;
typedef __attribute__((ext_vector_type(8)))  float    v8f;

#define B_   2
#define T_   16
#define N_   512
#define D_   256
#define H_   8
#define DH_  32
#define DP_  16
#define MR_  8
#define MM_  32
#define ROWS (B_*T_*N_)   /* 16384 */

// ---------------------------------------------------------------------------
// 1) Convert weight matrices f32 -> f16 (contraction dim contiguous per row)
// ---------------------------------------------------------------------------
__global__ void k_wconvert(const float* __restrict__ wq, const float* __restrict__ wk,
                           const float* __restrict__ wv, const float* __restrict__ wo,
                           _Float16* __restrict__ wq16, _Float16* __restrict__ wk16,
                           _Float16* __restrict__ wv16, _Float16* __restrict__ wo16) {
  int i = blockIdx.x * blockDim.x + threadIdx.x;   // 65536
  wq16[i] = (_Float16)wq[i];
  wk16[i] = (_Float16)wk[i];
  wv16[i] = (_Float16)wv[i];
  wo16[i] = (_Float16)wo[i];
}

// ---------------------------------------------------------------------------
// 2) RMSNorm: one wave per row of D=256, output f16
// ---------------------------------------------------------------------------
__global__ void k_rmsnorm(const float* __restrict__ x, const float* __restrict__ w,
                          _Float16* __restrict__ xn) {
  int wid = threadIdx.x >> 5, lane = threadIdx.x & 31;
  int row = blockIdx.x * 8 + wid;                  // 16384 rows
  const float* xr = x + (size_t)row * D_;
  float v[8]; float ss = 0.f;
  #pragma unroll
  for (int i = 0; i < 8; i++) { v[i] = xr[lane + i * 32]; ss += v[i] * v[i]; }
  #pragma unroll
  for (int off = 1; off < 32; off <<= 1) ss += __shfl_xor(ss, off, 32);
  float r = rsqrtf(ss * (1.0f / D_) + 1e-6f);
  _Float16* o = xn + (size_t)row * D_;
  #pragma unroll
  for (int i = 0; i < 8; i++) { int e = lane + i * 32; o[e] = (_Float16)(v[i] * r * w[e]); }
}

// ---------------------------------------------------------------------------
// 3) Per-point precompute: rotary sin/cos (first MR freqs) and tail_hg
// ---------------------------------------------------------------------------
__global__ void k_point(const float* __restrict__ pos, const float* __restrict__ dirs,
                        const float* __restrict__ freqs, const float* __restrict__ omega,
                        const float* __restrict__ pcw, const float* __restrict__ gain,
                        float* __restrict__ sc, float* __restrict__ tailhg) {
  int p = blockIdx.x * blockDim.x + threadIdx.x;   // B*N = 1024
  if (p >= B_ * N_) return;
  float px = pos[p*3+0], py = pos[p*3+1], pz = pos[p*3+2];
  float* scp = sc + (size_t)p * 16;
  #pragma unroll
  for (int f = 0; f < MR_; f++) {
    float a = (px*dirs[f*3+0] + py*dirs[f*3+1] + pz*dirs[f*3+2]) * freqs[f];
    scp[f]     = cosf(a);
    scp[8 + f] = sinf(a);
  }
  float tl[16];
  #pragma unroll
  for (int i = 0; i < 16; i++) tl[i] = 0.f;
  for (int m = 0; m < MM_; m++) {
    float pr = px*omega[m*3+0] + py*omega[m*3+1] + pz*omega[m*3+2];
    float c = cosf(pr), s = sinf(pr);
    #pragma unroll
    for (int dp = 0; dp < 16; dp++)
      tl[dp] += c * pcw[dp*64 + m] + s * pcw[dp*64 + 32 + m];
  }
  const float scale = 0.31622776601683794f;  // sqrt(0.1)
  for (int h = 0; h < H_; h++)
    #pragma unroll
    for (int dp = 0; dp < 16; dp++)
      tailhg[((size_t)p*H_ + h)*16 + dp] = scale * tl[dp] * gain[h*16 + dp];
}

// ---------------------------------------------------------------------------
// 4) Masks: additive key mask and query mask (valid & has_any_key)
// ---------------------------------------------------------------------------
__global__ void k_masks(const int* __restrict__ padm, const int* __restrict__ spikem,
                        float* __restrict__ maskadd, float* __restrict__ qmaskf) {
  int bt = blockIdx.x;            // 0..31
  int b  = bt / T_;
  int n  = threadIdx.x;           // 0..511
  int valid = padm[b*N_ + n] != 0;
  int spk   = (spikem[(size_t)bt*N_ + n] != 0) && valid;
  maskadd[(size_t)bt*N_ + n] = spk ? 0.0f : -1e30f;
  __shared__ int anyk;
  if (n == 0) anyk = 0;
  __syncthreads();
  if (spk) atomicOr(&anyk, 1);
  __syncthreads();
  qmaskf[(size_t)bt*N_ + n] = (valid && anyk) ? 1.0f : 0.0f;
}

// ---------------------------------------------------------------------------
// 5) QKV projection GEMM (WMMA). Only the first 16 columns of each 32-wide
//    head are needed (tail gets replaced / zeroed), so col tile ct covers
//    e = ct*32 + 0..15. One wave computes one 16x16 output tile, K-loop = 8.
// ---------------------------------------------------------------------------
__global__ void k_qkv(const _Float16* __restrict__ xn,
                      const _Float16* __restrict__ wq, const _Float16* __restrict__ wk,
                      const _Float16* __restrict__ wv,
                      float* __restrict__ qraw, float* __restrict__ kraw,
                      float* __restrict__ vraw) {
  int wid = threadIdx.x >> 5, lane = threadIdx.x & 31;
  int l = lane & 15, hi = lane >> 4;
  int rt = blockIdx.x * 8 + wid;   // row tile 0..1023
  int ct = blockIdx.y;             // head 0..7
  const _Float16* W = (blockIdx.z == 0) ? wq : (blockIdx.z == 1) ? wk : wv;
  float* out = (blockIdx.z == 0) ? qraw : (blockIdx.z == 1) ? kraw : vraw;
  const _Float16* arow = xn + (size_t)(rt * 16 + l) * D_;
  const _Float16* brow = W  + (size_t)(ct * 32 + l) * D_;
  __builtin_prefetch(arow, 0, 3);      // global_prefetch_b8
  __builtin_prefetch(brow, 0, 3);
  v8f acc = {};
  #pragma unroll
  for (int kk = 0; kk < 8; kk++) {
    int k0 = kk * 32 + hi * 16;
    v16h a = *(const v16h*)(arow + k0);
    v16h b = *(const v16h*)(brow + k0);
    acc = __builtin_amdgcn_wmma_f32_16x16x32_f16(false, a, false, b, (short)0, acc, false, false);
  }
  int colbase = ct * 16;
  #pragma unroll
  for (int j = 0; j < 8; j++) {
    int rrow = rt * 16 + j + hi * 8;
    out[(size_t)rrow * 128 + colbase + l] = acc[j];
  }
}

// ---------------------------------------------------------------------------
// 6) Post: rotary on the 16 kept dims, append tail_hg, pack attention layouts
//    qh/kh: (B,T,H,N,32) f16 ; vT: (B,T,H,16,N) f16 (transposed for B-frag)
// ---------------------------------------------------------------------------
__global__ void k_post(const float* __restrict__ qraw, const float* __restrict__ kraw,
                       const float* __restrict__ vraw, const float* __restrict__ sc,
                       const float* __restrict__ tailhg,
                       _Float16* __restrict__ qh, _Float16* __restrict__ kh,
                       _Float16* __restrict__ vT) {
  int gid = blockIdx.x * blockDim.x + threadIdx.x;  // B*T*N*H = 131072
  int h  = gid & 7;
  int n  = (gid >> 3) & (N_ - 1);
  int bt = gid >> 12;
  int b  = bt >> 4;
  size_t rawrow = (size_t)bt * N_ + n;
  const float* scp = sc + ((size_t)b * N_ + n) * 16;
  const float* tlp = tailhg + (((size_t)b * N_ + n) * H_ + h) * 16;
  float qv[16], kv[16];
  #pragma unroll
  for (int i = 0; i < 16; i++) {
    qv[i] = qraw[rawrow * 128 + h * 16 + i];
    kv[i] = kraw[rawrow * 128 + h * 16 + i];
  }
  _Float16 qo[32], ko[32];
  #pragma unroll
  for (int i = 0; i < 8; i++) {
    float c = scp[i], s = scp[8 + i];
    qo[2*i]   = (_Float16)(qv[2*i] * c - qv[2*i+1] * s);
    qo[2*i+1] = (_Float16)(qv[2*i] * s + qv[2*i+1] * c);
    ko[2*i]   = (_Float16)(kv[2*i] * c - kv[2*i+1] * s);
    ko[2*i+1] = (_Float16)(kv[2*i] * s + kv[2*i+1] * c);
  }
  #pragma unroll
  for (int dp = 0; dp < 16; dp++) {
    _Float16 tv = (_Float16)tlp[dp];
    qo[16 + dp] = tv;
    ko[16 + dp] = tv;
  }
  size_t bth = (size_t)bt * H_ + h;
  _Float16* qd = qh + (bth * N_ + n) * 32;
  _Float16* kd = kh + (bth * N_ + n) * 32;
  #pragma unroll
  for (int i = 0; i < 32; i++) { qd[i] = qo[i]; kd[i] = ko[i]; }
  #pragma unroll
  for (int d = 0; d < 16; d++)
    vT[(bth * 16 + d) * N_ + n] = (_Float16)vraw[rawrow * 128 + h * 16 + d];
}

// ---------------------------------------------------------------------------
// 7) Flash attention per (b,t,h). Block = 8 waves, each wave owns a 16-row
//    query strip; k (512x32), vT (16x512) and mask staged into LDS with the
//    CDNA5 async-to-LDS path (GLOBAL_LOAD_ASYNC_TO_LDS_B128, ASYNCcnt),
//    bypassing VGPR staging. K=Dh=32 exactly, so each 16x16 score tile is one
//    WMMA. Online softmax; P converts C-layout -> A-fragment via per-wave LDS.
// ---------------------------------------------------------------------------
__global__ void __launch_bounds__(256) k_attn(const _Float16* __restrict__ qh,
                                              const _Float16* __restrict__ kh,
                                              const _Float16* __restrict__ vT,
                                              const float* __restrict__ maskadd,
                                              const float* __restrict__ qmaskf,
                                              _Float16* __restrict__ ao) {
  __shared__ __align__(16) _Float16 ksh[N_ * 32];      // 32 KB
  __shared__ __align__(16) _Float16 vsh[16 * N_];      // 16 KB
  __shared__ __align__(16) _Float16 psh[8][16 * 32];   //  8 KB
  __shared__ __align__(16) float    msh[N_];           //  2 KB

  int bth = blockIdx.x >> 2, sb = blockIdx.x & 3;
  int bt = bth >> 3;
  int tid = threadIdx.x;

  // --- async copy global -> LDS (no VGPR staging), tracked by ASYNCcnt ---
  {
    const char* kg = (const char*)(kh + (size_t)bth * N_ * 32);   // 32 KB
    #pragma unroll
    for (int c = 0; c < 8; c++) {
      int idx = tid + c * 256;                                    // 16B chunks
      unsigned lds = (unsigned)(size_t)((char*)ksh + idx * 16);
      unsigned long long ga = (unsigned long long)(size_t)(kg + idx * 16);
      asm volatile("global_load_async_to_lds_b128 %0, %1, off"
                   :: "v"(lds), "v"(ga) : "memory");
    }
    const char* vg = (const char*)(vT + (size_t)bth * 16 * N_);   // 16 KB
    #pragma unroll
    for (int c = 0; c < 4; c++) {
      int idx = tid + c * 256;
      unsigned lds = (unsigned)(size_t)((char*)vsh + idx * 16);
      unsigned long long ga = (unsigned long long)(size_t)(vg + idx * 16);
      asm volatile("global_load_async_to_lds_b128 %0, %1, off"
                   :: "v"(lds), "v"(ga) : "memory");
    }
    const char* mg = (const char*)(maskadd + (size_t)bt * N_);    // 2 KB
    if (tid < 128) {
      unsigned lds = (unsigned)(size_t)((char*)msh + tid * 16);
      unsigned long long ga = (unsigned long long)(size_t)(mg + tid * 16);
      asm volatile("global_load_async_to_lds_b128 %0, %1, off"
                   :: "v"(lds), "v"(ga) : "memory");
    }
    asm volatile("s_wait_asynccnt 0x0" ::: "memory");
  }
  __syncthreads();

  int wid = tid >> 5, lane = tid & 31, l = lane & 15, hi = lane >> 4;
  int strip = sb * 128 + wid * 16;

  v16h qf = *(const v16h*)(qh + ((size_t)bth * N_ + strip + l) * 32 + hi * 16);
  v8f acc = {};
  float mrow[8], lrow[8];
  #pragma unroll
  for (int j = 0; j < 8; j++) { mrow[j] = -3e38f; lrow[j] = 0.f; }
  const float scale = 0.17677669529663687f;   // 1/sqrt(Dh)
  _Float16* pbuf = psh[wid];

  for (int jj = 0; jj < 16; jj++) {
    int kb = jj * 32;
    v16h kf0 = *(const v16h*)(ksh + (kb + l) * 32 + hi * 16);
    v16h kf1 = *(const v16h*)(ksh + (kb + 16 + l) * 32 + hi * 16);
    v8f z = {};
    v8f S0 = __builtin_amdgcn_wmma_f32_16x16x32_f16(false, qf, false, kf0, (short)0, z, false, false);
    v8f S1 = __builtin_amdgcn_wmma_f32_16x16x32_f16(false, qf, false, kf1, (short)0, z, false, false);
    float m0 = msh[kb + l], m1 = msh[kb + 16 + l];
    #pragma unroll
    for (int j = 0; j < 8; j++) {
      float s0 = S0[j] * scale + m0;
      float s1 = S1[j] * scale + m1;
      float tmax = fmaxf(s0, s1);
      #pragma unroll
      for (int off = 1; off < 16; off <<= 1) tmax = fmaxf(tmax, __shfl_xor(tmax, off, 32));
      float nm = fmaxf(mrow[j], tmax);
      float corr = __expf(mrow[j] - nm);
      float p0 = __expf(s0 - nm), p1 = __expf(s1 - nm);
      float rs = p0 + p1;
      #pragma unroll
      for (int off = 1; off < 16; off <<= 1) rs += __shfl_xor(rs, off, 32);
      lrow[j] = lrow[j] * corr + rs;
      mrow[j] = nm;
      acc[j] *= corr;
      int prow = j + hi * 8;
      pbuf[prow * 32 + l]      = (_Float16)p0;
      pbuf[prow * 32 + 16 + l] = (_Float16)p1;
    }
    asm volatile("s_wait_dscnt 0x0" ::: "memory");
    v16h pf = *(const v16h*)(pbuf + l * 32 + hi * 16);
    v16h vf = *(const v16h*)(vsh + l * N_ + kb + hi * 16);
    acc = __builtin_amdgcn_wmma_f32_16x16x32_f16(false, pf, false, vf, (short)0, acc, false, false);
  }

  int h = bth & 7;
  #pragma unroll
  for (int j = 0; j < 8; j++) {
    int n = strip + j + hi * 8;
    float inv = lrow[j] > 0.f ? 1.0f / lrow[j] : 0.f;
    float qm = qmaskf[(size_t)bt * N_ + n];
    float val = acc[j] * inv * qm;
    size_t base = ((size_t)bt * N_ + n) * D_ + h * 32;
    ao[base + l]      = (_Float16)val;
    ao[base + 16 + l] = (_Float16)0.f;    // zeroed tail dims
  }
}

// ---------------------------------------------------------------------------
// 8) Output projection GEMM + residual: y = ao @ Wo^T + x
// ---------------------------------------------------------------------------
__global__ void k_out(const _Float16* __restrict__ ao, const _Float16* __restrict__ wo,
                      const float* __restrict__ x, float* __restrict__ y) {
  int wid = threadIdx.x >> 5, lane = threadIdx.x & 31;
  int l = lane & 15, hi = lane >> 4;
  int rt = blockIdx.x * 8 + wid;  // 0..1023
  int ct = blockIdx.y;            // 0..15
  const _Float16* arow = ao + (size_t)(rt * 16 + l) * D_;
  const _Float16* brow = wo + (size_t)(ct * 16 + l) * D_;
  __builtin_prefetch(arow, 0, 3);
  __builtin_prefetch(brow, 0, 3);
  v8f acc = {};
  #pragma unroll
  for (int kk = 0; kk < 8; kk++) {
    int k0 = kk * 32 + hi * 16;
    v16h a = *(const v16h*)(arow + k0);
    v16h b = *(const v16h*)(brow + k0);
    acc = __builtin_amdgcn_wmma_f32_16x16x32_f16(false, a, false, b, (short)0, acc, false, false);
  }
  #pragma unroll
  for (int j = 0; j < 8; j++) {
    size_t idx = (size_t)(rt * 16 + j + hi * 8) * D_ + ct * 16 + l;
    y[idx] = acc[j] + x[idx];
  }
}

// ---------------------------------------------------------------------------
extern "C" void kernel_launch(void* const* d_in, const int* in_sizes, int n_in,
                              void* d_out, int out_size, void* d_ws, size_t ws_size,
                              hipStream_t stream) {
  const float* x     = (const float*)d_in[0];
  const float* ppos  = (const float*)d_in[1];
  const int*   padm  = (const int*)d_in[2];
  const int*   spkm  = (const int*)d_in[3];
  const float* rmsw  = (const float*)d_in[4];
  const float* Wq    = (const float*)d_in[5];
  const float* Wk    = (const float*)d_in[6];
  const float* Wv    = (const float*)d_in[7];
  const float* Wo    = (const float*)d_in[8];
  const float* rdirs = (const float*)d_in[9];
  const float* rfreq = (const float*)d_in[10];
  const float* omeg  = (const float*)d_in[11];
  const float* pcw   = (const float*)d_in[12];
  const float* gain  = (const float*)d_in[13];
  float* y = (float*)d_out;

  char* p = (char*)d_ws;
  auto alloc = [&](size_t bytes) -> char* {
    char* r = p;
    p += (bytes + 255) & ~(size_t)255;
    return r;
  };
  _Float16* xn16   = (_Float16*)alloc((size_t)ROWS * D_ * 2);         // 8 MB
  _Float16* wq16   = (_Float16*)alloc((size_t)D_ * D_ * 2);
  _Float16* wk16   = (_Float16*)alloc((size_t)D_ * D_ * 2);
  _Float16* wv16   = (_Float16*)alloc((size_t)D_ * D_ * 2);
  _Float16* wo16   = (_Float16*)alloc((size_t)D_ * D_ * 2);
  float*    qraw   = (float*)alloc((size_t)ROWS * 128 * 4);           // 8 MB
  float*    kraw   = (float*)alloc((size_t)ROWS * 128 * 4);
  float*    vraw   = (float*)alloc((size_t)ROWS * 128 * 4);
  float*    sc     = (float*)alloc((size_t)B_ * N_ * 16 * 4);
  float*    tailhg = (float*)alloc((size_t)B_ * N_ * H_ * 16 * 4);
  float*    maskad = (float*)alloc((size_t)B_ * T_ * N_ * 4);
  float*    qmaskf = (float*)alloc((size_t)B_ * T_ * N_ * 4);
  _Float16* qh     = (_Float16*)alloc((size_t)B_ * T_ * H_ * N_ * 32 * 2);  // 8 MB
  _Float16* kh     = (_Float16*)alloc((size_t)B_ * T_ * H_ * N_ * 32 * 2);  // 8 MB
  _Float16* vT     = (_Float16*)alloc((size_t)B_ * T_ * H_ * 16 * N_ * 2);  // 4 MB
  _Float16* ao     = (_Float16*)alloc((size_t)ROWS * D_ * 2);               // 8 MB

  k_wconvert<<<dim3(D_ * D_ / 256), dim3(256), 0, stream>>>(Wq, Wk, Wv, Wo,
                                                            wq16, wk16, wv16, wo16);
  k_rmsnorm<<<dim3(ROWS / 8), dim3(256), 0, stream>>>(x, rmsw, xn16);
  k_point<<<dim3((B_ * N_ + 255) / 256), dim3(256), 0, stream>>>(ppos, rdirs, rfreq,
                                                                 omeg, pcw, gain, sc, tailhg);
  k_masks<<<dim3(B_ * T_), dim3(N_), 0, stream>>>(padm, spkm, maskad, qmaskf);
  k_qkv<<<dim3(ROWS / 16 / 8, 8, 3), dim3(256), 0, stream>>>(xn16, wq16, wk16, wv16,
                                                             qraw, kraw, vraw);
  k_post<<<dim3(B_ * T_ * N_ * H_ / 256), dim3(256), 0, stream>>>(qraw, kraw, vraw, sc,
                                                                  tailhg, qh, kh, vT);
  k_attn<<<dim3(B_ * T_ * H_ * 4), dim3(256), 0, stream>>>(qh, kh, vT, maskad, qmaskf, ao);
  k_out<<<dim3(ROWS / 16 / 8, D_ / 16), dim3(256), 0, stream>>>(ao, wo16, x, y);
}
